// MatrixAttention_25958782337856
// MI455X (gfx1250) — matrix operand order repaired and checked by reading, not compiled, not run on hardware
//
#include <hip/hip_runtime.h>

// ---------------------------------------------------------------------------
// MatrixAttention dot-product similarity == batched GEMM C[b] = A[b] * M2[b]^T
//   A  : [16][1024][256] fp32    M2 : [16][1024][256] fp32
//   C  : [16][1024][1024] fp32
// Strategy (MI455X / gfx1250, wave32):
//   * bf16x3 split-precision GEMM on V_WMMA_F32_16X16X32_BF16 (fp32 accum):
//       x = hi(x) + lo(x);  C += Ahi*Bhi + Ahi*Blo + Alo*Bhi
//   * f32 -> bf16 hi/lo conversion ONCE per element at staging; LDS holds
//     pre-swizzled bf16 fragments; compute loop = ds_load_b128 + v_wmma only.
//   * Split barrier with LDS-only fences (waits dscnt only), so the chunk
//     k+1 HBM prefetch issued just before the barrier stays in flight and
//     overlaps chunk k's WMMAs.  Single prefetch register bank: the awaited
//     loads are one full compute-phase old, and one bank avoids the VGPR
//     spills a second bank provoked.
//   * 128x128 C-tile per 256-thread block, 8 waves each computing 64x32.
// ---------------------------------------------------------------------------

#define R1n   1024
#define R2n   1024
#define DDn   256
#define NBn   16

#define BM    128
#define BN    128
#define KCn   32
#define SB2   40          // LDS row stride in bf16 elems (80B: 16B-aligned, padded)
#define NTHR  256
#define NKC   (DDn / KCn) // 8 K-chunks

typedef __attribute__((ext_vector_type(16))) __bf16 v16bf;
typedef __attribute__((ext_vector_type(8)))  float  v8f;

#if defined(__HIP_DEVICE_COMPILE__) && __has_builtin(__builtin_amdgcn_s_barrier_signal) && \
    __has_builtin(__builtin_amdgcn_s_barrier_wait)
#define HAVE_SPLIT_BAR 1
#else
#define HAVE_SPLIT_BAR 0
#endif

// Workgroup barrier that publishes/acquires ONLY LDS (waits dscnt, not loadcnt),
// so in-flight global prefetch loads survive across it.
__device__ __forceinline__ void wg_barrier_lds() {
#if HAVE_SPLIT_BAR
  __builtin_amdgcn_fence(__ATOMIC_RELEASE, "workgroup", "local");
  __builtin_amdgcn_s_barrier_signal(-1);
  __builtin_amdgcn_s_barrier_wait(-1);
  __builtin_amdgcn_fence(__ATOMIC_ACQUIRE, "workgroup", "local");
#else
  __syncthreads();
#endif
}

// Split one f32 4-vector into packed bf16 hi/lo 4-vectors and store (8B each).
__device__ __forceinline__ void split_store4(__bf16* dh, __bf16* dl, float4 v) {
  union { __bf16 b[4]; uint2 u; } H, L;
  float c[4] = {v.x, v.y, v.z, v.w};
#pragma unroll
  for (int k = 0; k < 4; ++k) {
    __bf16 h = (__bf16)c[k];
    H.b[k] = h;
    L.b[k] = (__bf16)(c[k] - (float)h);   // residual -> v_fma_mix + v_cvt_pk
  }
  *(uint2*)dh = H.u;
  *(uint2*)dl = L.u;
}

// Load one 32B (16 x bf16) fragment from LDS (two ds_load_b128).
__device__ __forceinline__ v16bf fragld(const __bf16* p) {
  union { v16bf v; uint4 q[2]; } u;
  const uint4* s = (const uint4*)p;
  u.q[0] = s[0];
  u.q[1] = s[1];
  return u.v;
}

__global__ __launch_bounds__(NTHR)
void matattn_bmm_nt_bf16x3(const float* __restrict__ A,
                           const float* __restrict__ M2,
                           float* __restrict__ C) {
  // bf16 hi/lo tiles, double buffered: 4 * 2 * 128*40 * 2B = 80 KB
  __shared__ __align__(16) __bf16 sAh[2][BM * SB2];
  __shared__ __align__(16) __bf16 sAl[2][BM * SB2];
  __shared__ __align__(16) __bf16 sBh[2][BN * SB2];
  __shared__ __align__(16) __bf16 sBl[2][BN * SB2];

  const int b  = blockIdx.z;
  const int mt = blockIdx.y;
  const int nt = blockIdx.x;

  const float* Ab = A  + ((size_t)b * R1n + (size_t)mt * BM) * DDn;
  const float* Bb = M2 + ((size_t)b * R2n + (size_t)nt * BN) * DDn;

  const int t    = threadIdx.x;
  const int lane = t & 31;
  const int wave = t >> 5;
  const int wm   = wave >> 2;            // 0..1 : 64-row slice
  const int wn   = wave & 3;             // 0..3 : 32-col slice
  const int lr   = lane & 15;
  const int koff = (lane >> 4) << 4;     // 0 or 16 bf16 elems into the row

  v8f acc[4][2];
#pragma unroll
  for (int mi = 0; mi < 4; ++mi)
#pragma unroll
    for (int ni = 0; ni < 2; ++ni)
      acc[mi][ni] = (v8f){0.f, 0.f, 0.f, 0.f, 0.f, 0.f, 0.f, 0.f};

  float4 pa[4], pb[4];   // single-bank in-flight global data for next chunk

  // Issue global f32 loads for chunk kc (8 x global_load_b128 per thread).
  auto gload = [&](int kc) {
#pragma unroll
    for (int i = 0; i < 4; ++i) {
      const int q   = t + NTHR * i;      // 0..1023 = 128 rows x 8 K-groups
      const int row = q >> 3;
      const int g   = q & 7;             // 4-wide K group within the 32 chunk
      pa[i] = *(const float4*)(Ab + (size_t)row * DDn + kc * KCn + g * 4);
      pb[i] = *(const float4*)(Bb + (size_t)row * DDn + kc * KCn + g * 4);
    }
  };

  // Convert held f32 data to bf16 hi/lo and store swizzled into LDS buffer.
  // A and B rows reordered [K0-7 | K16-23 | K8-15 | K24-31] so a lane's 16 fragment
  // elements (ISA split-K layout) are one contiguous 32B read.
  auto lstore = [&](int bufsel) {
#pragma unroll
    for (int i = 0; i < 4; ++i) {
      const int q   = t + NTHR * i;
      const int row = q >> 3;
      const int g   = q & 7;
      const int pA  = (g & 1) | ((g & 4) >> 1) | ((g & 2) << 1); // swap bits 1<->2
      split_store4(&sAh[bufsel][row * SB2 + pA * 4],
                   &sAl[bufsel][row * SB2 + pA * 4], pa[i]);
      split_store4(&sBh[bufsel][row * SB2 + pA * 4],
                   &sBl[bufsel][row * SB2 + pA * 4], pb[i]);
    }
  };

  gload(0);

  for (int kc = 0; kc < NKC; ++kc) {
    const int buf = kc & 1;
    lstore(buf);                      // convert chunk k (loads are a phase old)
    if (kc + 1 < NKC) gload(kc + 1);  // issue next HBM loads; they stay in
    wg_barrier_lds();                 // flight across this dscnt-only barrier

    // B fragments for this wave's 32 columns (reused across 4 M tiles)
    v16bf bh[2], bl[2];
#pragma unroll
    for (int ni = 0; ni < 2; ++ni) {
      const int roff = (wn * 32 + ni * 16 + lr) * SB2 + koff;
      bh[ni] = fragld(&sBh[buf][roff]);
      bl[ni] = fragld(&sBl[buf][roff]);
    }

#pragma unroll
    for (int mi = 0; mi < 4; ++mi) {
      const int roff = (wm * 64 + mi * 16 + lr) * SB2 + koff;
      v16bf ah = fragld(&sAh[buf][roff]);
      v16bf al = fragld(&sAl[buf][roff]);
#pragma unroll
      for (int ni = 0; ni < 2; ++ni) {
        acc[mi][ni] = __builtin_amdgcn_wmma_f32_16x16x32_bf16(
            false, al, false, bh[ni], (short)0, acc[mi][ni], false, false);
        acc[mi][ni] = __builtin_amdgcn_wmma_f32_16x16x32_bf16(
            false, ah, false, bl[ni], (short)0, acc[mi][ni], false, false);
        acc[mi][ni] = __builtin_amdgcn_wmma_f32_16x16x32_bf16(
            false, ah, false, bh[ni], (short)0, acc[mi][ni], false, false);
      }
    }
    wg_barrier_lds();  // all waves done reading `buf` before it is restaged
  }

  // Epilogue: C/D layout -> VGPR j holds M = j + 8*(lane>>4), N = lane&15
  float* Cb = C + ((size_t)b << 20) + (size_t)(mt * BM) * R2n + (size_t)(nt * BN);
  const int rsub = (lane >> 4) << 3;
#pragma unroll
  for (int mi = 0; mi < 4; ++mi)
#pragma unroll
    for (int ni = 0; ni < 2; ++ni)
#pragma unroll
      for (int j = 0; j < 8; ++j) {
        const int r = wm * 64 + mi * 16 + rsub + j;
        const int c = wn * 32 + ni * 16 + lr;
        Cb[(size_t)r * R2n + c] = acc[mi][ni][j];
      }
}

extern "C" void kernel_launch(void* const* d_in, const int* in_sizes, int n_in,
                              void* d_out, int out_size, void* d_ws, size_t ws_size,
                              hipStream_t stream) {
  const float* A  = (const float*)d_in[0];
  const float* M2 = (const float*)d_in[1];
  float* C = (float*)d_out;
  dim3 grid(R2n / BN, R1n / BM, NBn);   // (8, 8, 16) workgroups
  matattn_bmm_nt_bf16x3<<<grid, dim3(NTHR), 0, stream>>>(A, M2, C);
}
